// NestedGIN_eff_65798898974933
// MI455X (gfx1250) — matrix operand order repaired and checked by reading, not compiled, not run on hardware
//
#include <hip/hip_runtime.h>
#include <hip/hip_bf16.h>

// ---------------------------------------------------------------------------
// NestedGIN pipeline for MI455X (gfx1250, wave32).
// GEMMs: f16 inputs, f32 accumulation via v_wmma_f32_16x16x32_f16.
//   - weights pre-swizzled to WMMA fragment-major layout -> B frags are wide
//     ds_load_b128 pairs (no scalar u16 assembly)
//   - kc-outer / nt-inner WMMA order: adjacent WMMAs hit different
//     accumulators (no D-operand RAW chains)
//   - async global->LDS weight staging (global_load_async_to_lds_b128)
// Scatter / BN / pooling: fp32 with float4 traffic + atomics.
// ---------------------------------------------------------------------------

typedef _Float16 h16;
typedef __attribute__((ext_vector_type(16))) _Float16 v16h;
typedef __attribute__((ext_vector_type(8)))  _Float16 v8h;
typedef __attribute__((ext_vector_type(8)))  float    v8f;

#define NN   50000
#define EE   800000
#define PP   2400000
#define HH   128
#define GG   512
#define ZIN  1800
#define OUTD 128

#if __has_builtin(__builtin_amdgcn_global_load_async_to_lds_b128)
#define ASYNC_LDS 1
// builtin parameter types (from hipcc diagnostic): v4i in AS1 / AS3
typedef int v4i_vs __attribute__((vector_size(16)));
typedef __attribute__((address_space(1))) v4i_vs* gv4p;
typedef __attribute__((address_space(3))) v4i_vs* lv4p;
#endif

__device__ __forceinline__ void wait_async_zero() {
#if __has_builtin(__builtin_amdgcn_s_wait_asynccnt)
  __builtin_amdgcn_s_wait_asynccnt(0);
#else
  asm volatile("s_wait_asynccnt 0x0" ::: "memory");
#endif
}

// fragment-major swizzled weight layout (per 128-K chunk):
//   32 fragments (nt 0..7 x kc 0..3), each 32 lanes x 16 halves,
//   lane slot skewed by +8 halves every 8 lanes to spread LDS banks.
#define WS_FRAG_STRIDE 544                    // halves per fragment (32*16 + 4*8)
#define WS_HALVES      (32 * WS_FRAG_STRIDE)  // 17408 halves = 34816 B per chunk

__device__ __forceinline__ int lane_slot(int l) { return l * 16 + (l >> 3) * 8; }

// -------------------------------- utility kernels --------------------------

__global__ void k_zero_f32(float* __restrict__ p, long long n) {
  long long i = (long long)blockIdx.x * blockDim.x + threadIdx.x;
  long long stride = (long long)gridDim.x * blockDim.x;
  for (; i < n; i += stride) p[i] = 0.f;
}

__global__ void k_f32_to_f16(const float* __restrict__ s, h16* __restrict__ d, long long n) {
  long long i = (long long)blockIdx.x * blockDim.x + threadIdx.x;
  long long stride = (long long)gridDim.x * blockDim.x;
  for (; i < n; i += stride) d[i] = (h16)s[i];
}

// convert f32 weights [K x 128] into f16 fragment-major swizzled layout
__global__ void k_w_swizzle(const float* __restrict__ W, h16* __restrict__ out, int K) {
  long long i = (long long)blockIdx.x * blockDim.x + threadIdx.x;
  if (i >= (long long)K * HH) return;
  int k = (int)(i >> 7), n = (int)(i & 127);
  int kb    = k >> 7;        // 128-K chunk index
  int kl128 = k & 127;
  int kc    = kl128 >> 5;    // 32-K sub-chunk
  int kl    = kl128 & 31;
  int hlf   = kl >> 4;
  int t     = kl & 15;       // position within lane's 16 halves
  int nt    = n >> 4;
  int lane  = hlf * 16 + (n & 15);
  long long dst = (long long)kb * WS_HALVES +
                  (long long)(nt * 4 + kc) * WS_FRAG_STRIDE + lane_slot(lane) + t;
  out[dst] = (h16)W[i];
}

// z[pos_batch[p], :] += z_initial[pos_index[p], :] * pos_enc[p]
__global__ void k_scatter_pos(const float* __restrict__ zi, const float* __restrict__ pe,
                              const int* __restrict__ pidx, const int* __restrict__ pbatch,
                              float* __restrict__ z, int P) {
  int p = blockIdx.x;
  if (p >= P) return;
  int c = threadIdx.x;                       // 128 threads
  float w = pe[p];
  float v = zi[(size_t)pidx[p] * HH + c] * w;
  atomicAdd(&z[(size_t)pbatch[p] * HH + c], v);
}

// per-channel sum / sumsq over rows (x is [rows,128])
__global__ void k_bn_stats(const float* __restrict__ x, int rows, float* __restrict__ stats) {
  int c = threadIdx.x;                       // 128 threads
  float s = 0.f, q = 0.f;
  for (int r = blockIdx.x; r < rows; r += gridDim.x) {
    float v = x[(size_t)r * HH + c];
    s += v; q += v * v;
  }
  atomicAdd(&stats[c], s);
  atomicAdd(&stats[HH + c], q);
}

__global__ void k_bn_finalize(const float* __restrict__ stats, const float* __restrict__ g,
                              const float* __restrict__ be, float inv_rows, float* __restrict__ ss) {
  int c = threadIdx.x;                       // 128 threads
  float m  = stats[c] * inv_rows;
  float v  = stats[HH + c] * inv_rows - m * m;
  float is = rsqrtf(v + 1e-5f);
  float sc = g[c] * is;
  ss[c]      = sc;
  ss[HH + c] = be[c] - m * sc;
}

// val = relu(y*scale+shift); optional f32 store (strided, accumulating) + f16 store
__global__ void k_bn_relu_apply(const float* __restrict__ y, const float* __restrict__ ss,
                                float* __restrict__ out32, int ldo, int accumulate,
                                h16* __restrict__ out16, int rows) {
  long long i = (long long)blockIdx.x * blockDim.x + threadIdx.x;
  long long total = (long long)rows * HH;
  long long stride = (long long)gridDim.x * blockDim.x;
  for (; i < total; i += stride) {
    int r = (int)(i >> 7), c = (int)(i & 127);
    float val = fmaxf(0.f, y[i] * ss[c] + ss[HH + c]);
    if (out32) {
      float* p = out32 + (size_t)r * ldo + c;
      if (accumulate) *p += val; else *p = val;
    }
    if (out16) out16[i] = (h16)val;
  }
}

// y[r,c] = u[r]*W1[c] + b1[c]   (rank-1 "GEMM" for din==1 layers)
__global__ void k_rank1(const float* __restrict__ u, const float* __restrict__ W1,
                        const float* __restrict__ b1, float* __restrict__ y, int rows) {
  long long i = (long long)blockIdx.x * blockDim.x + threadIdx.x;
  long long total = (long long)rows * HH;
  long long stride = (long long)gridDim.x * blockDim.x;
  for (; i < total; i += stride) {
    int r = (int)(i >> 7), c = (int)(i & 127);
    y[i] = u[r] * W1[c] + b1[c];
  }
}

__global__ void k_colsum(const float* __restrict__ W, float* __restrict__ cs) {
  int c = threadIdx.x;                       // 128 threads
  float s = 0.f;
  for (int k = 0; k < HH; ++k) s += W[(size_t)k * HH + c];
  cs[c] = s;
}

// aggr[dst] += relu(h[src]+elin) , 4 channels per thread
__global__ __launch_bounds__(256)
void k_edge_msg(const float* __restrict__ h, int ldh, const float* __restrict__ elin,
                const int* __restrict__ src, const int* __restrict__ dst,
                float* __restrict__ aggr, int E) {
  long long gid = (long long)blockIdx.x * blockDim.x + threadIdx.x;
  long long total = (long long)E * 32;
  if (gid >= total) return;
  int e = (int)(gid >> 5);
  int c4 = (int)(gid & 31) * 4;
  int s = src[e], d = dst[e];
  float4 ev = *(const float4*)(elin + (size_t)e * HH + c4);
  float4 hv = *(const float4*)(h + (size_t)s * ldh + c4);
  float* ap = aggr + (size_t)d * HH + c4;
  atomicAdd(ap + 0, fmaxf(0.f, hv.x + ev.x));
  atomicAdd(ap + 1, fmaxf(0.f, hv.y + ev.y));
  atomicAdd(ap + 2, fmaxf(0.f, hv.z + ev.z));
  atomicAdd(ap + 3, fmaxf(0.f, hv.w + ev.w));
}

// dual messages: elin and elin+colsum(W)  ((node_id+1)@W = node_id@W + colsum)
__global__ __launch_bounds__(256)
void k_edge_msg_dual(const float* __restrict__ h, int ldh, const float* __restrict__ elin,
                     const float* __restrict__ cs, const int* __restrict__ src,
                     const int* __restrict__ dst, float* __restrict__ aggrA,
                     float* __restrict__ aggrB, int E) {
  long long gid = (long long)blockIdx.x * blockDim.x + threadIdx.x;
  long long total = (long long)E * 32;
  if (gid >= total) return;
  int e = (int)(gid >> 5);
  int c4 = (int)(gid & 31) * 4;
  int s = src[e], d = dst[e];
  float4 ev = *(const float4*)(elin + (size_t)e * HH + c4);
  float4 hv = *(const float4*)(h + (size_t)s * ldh + c4);
  float4 cv = *(const float4*)(cs + c4);
  float* pa = aggrA + (size_t)d * HH + c4;
  float* pb = aggrB + (size_t)d * HH + c4;
  atomicAdd(pa + 0, fmaxf(0.f, hv.x + ev.x));
  atomicAdd(pa + 1, fmaxf(0.f, hv.y + ev.y));
  atomicAdd(pa + 2, fmaxf(0.f, hv.z + ev.z));
  atomicAdd(pa + 3, fmaxf(0.f, hv.w + ev.w));
  atomicAdd(pb + 0, fmaxf(0.f, hv.x + ev.x + cv.x));
  atomicAdd(pb + 1, fmaxf(0.f, hv.y + ev.y + cv.y));
  atomicAdd(pb + 2, fmaxf(0.f, hv.z + ev.z + cv.z));
  atomicAdd(pb + 3, fmaxf(0.f, hv.w + ev.w + cv.w));
}

// conv1 edge: scalar message m = relu(x[src] + dot(zemb[e],w) + b); wave-per-edge
__global__ __launch_bounds__(256)
void k_conv1_edge(const h16* __restrict__ zemb, const float* __restrict__ w,
                  const float* __restrict__ b, const float* __restrict__ x,
                  const int* __restrict__ src, const int* __restrict__ dst,
                  float* __restrict__ aggr, int E) {
  int wid  = blockIdx.x * 8 + (threadIdx.x >> 5);
  int lane = threadIdx.x & 31;
  if (wid >= E) return;
  float part = 0.f;
#pragma unroll
  for (int j = 0; j < 4; ++j) {
    int c = lane * 4 + j;
    part += (float)zemb[(size_t)wid * HH + c] * w[c];
  }
  for (int off = 16; off; off >>= 1) part += __shfl_xor(part, off);
  if (lane == 0) {
    float m = fmaxf(0.f, x[src[wid]] + part + b[0]);
    atomicAdd(&aggr[dst[wid]], m);
  }
}

// u16 = f16((1+eps)*h + aggr)
__global__ void k_combine_u(const float* __restrict__ h, int ldh, const float* __restrict__ aggr,
                            const float* __restrict__ epsp, h16* __restrict__ u16, int rows) {
  long long i = (long long)blockIdx.x * blockDim.x + threadIdx.x;
  long long total = (long long)rows * HH;
  long long stride = (long long)gridDim.x * blockDim.x;
  float oe = 1.f + epsp[0];
  for (; i < total; i += stride) {
    int r = (int)(i >> 7), c = (int)(i & 127);
    u16[i] = (h16)(oe * h[(size_t)r * ldh + c] + aggr[i]);
  }
}

__global__ void k_combine_scalar(const float* __restrict__ x, const float* __restrict__ aggr,
                                 const float* __restrict__ epsp, float* __restrict__ u, int n) {
  int i = blockIdx.x * blockDim.x + threadIdx.x;
  if (i < n) u[i] = (1.f + epsp[0]) * x[i] + aggr[i];
}

// mean pooling over batch
__global__ void k_pool_add(const float* __restrict__ cat, const int* __restrict__ batch,
                           float* __restrict__ pooled, float* __restrict__ cnt, int n) {
  int i = blockIdx.x;
  if (i >= n) return;
  int b = batch[i];
  for (int c = threadIdx.x; c < 512; c += blockDim.x)
    atomicAdd(&pooled[(size_t)b * 512 + c], cat[(size_t)i * 512 + c]);
  if (threadIdx.x == 0) atomicAdd(&cnt[b], 1.f);
}

__global__ void k_pool_div(float* __restrict__ pooled, const float* __restrict__ cnt,
                           h16* __restrict__ p16) {
  int i = blockIdx.x * blockDim.x + threadIdx.x;
  if (i >= GG * 512) return;
  int g = i >> 9;
  float v = pooled[i] / fmaxf(cnt[g], 1.f);
  pooled[i] = v;
  p16[i] = (h16)v;
}

// -------------------------------- WMMA GEMM --------------------------------
// C[M,128] = A[M,K]*Wswz[K,128] + bias ; A f16 row-major, W pre-swizzled ;
// C f32 (ldc). block = 128 threads (4 waves); each wave: 16-row strip x 8 tiles.
__global__ __launch_bounds__(128)
void k_gemm_wmma(const h16* __restrict__ A, const h16* __restrict__ Wswz,
                 const float* __restrict__ bias, float* __restrict__ C,
                 int M, int K, int ldc) {
  __shared__ h16 WsF[WS_HALVES];       // fragment-major weight chunk (34.0 KB)
  __shared__ h16 As[64][HH + 8];       // padded A tile (17.0 KB)
  const int tid   = threadIdx.x;
  const int wave  = tid >> 5;
  const int lane  = tid & 31;
  const int row16 = lane & 15;
  const int hlf   = lane >> 4;
  const int blockRow = blockIdx.x * 64;
  const int myslot = lane_slot(lane);

  v8f acc[8];
#pragma unroll
  for (int nt = 0; nt < 8; ++nt) {
    float bv = bias ? bias[nt * 16 + row16] : 0.f;
#pragma unroll
    for (int r = 0; r < 8; ++r) acc[nt][r] = bv;
  }

  for (int kb = 0; kb < K; kb += 128) {
    if (kb + 128 < K)  // pull next A chunk toward L2 early (global_prefetch_b8)
      __builtin_prefetch(A + (size_t)(blockRow + (tid & 63)) * K + kb + 128, 0, 1);

    // stage swizzled weight chunk: straight contiguous 16B copy
    {
      const h16* wsrc = Wswz + (size_t)(kb >> 7) * WS_HALVES;
#ifdef ASYNC_LDS
      for (int idx = tid; idx < WS_HALVES / 8; idx += 128) {
        __builtin_amdgcn_global_load_async_to_lds_b128(
            (gv4p)(wsrc + (size_t)idx * 8),
            (lv4p)(WsF + (size_t)idx * 8),
            0, 0);
      }
#else
      const uint4* ws4 = (const uint4*)wsrc;
      uint4* wdst = (uint4*)WsF;
      for (int idx = tid; idx < WS_HALVES / 8; idx += 128) wdst[idx] = ws4[idx];
#endif
    }
    // stage A tile [64 x 128] as v8h (zero-pad past M)
    for (int idx = tid; idx < 64 * 16; idx += 128) {
      int r = idx >> 4, cg = (idx & 15) * 8;
      int grow = blockRow + r;
      v8h av = {};
      if (grow < M) av = *(const v8h*)(A + (size_t)grow * K + kb + cg);
      *(v8h*)&As[r][cg] = av;
    }
#ifdef ASYNC_LDS
    wait_async_zero();
#endif
    __syncthreads();

    // A fragments: two contiguous 16B LDS loads per 32-K chunk, concat
    v16h afrag[4];
#pragma unroll
    for (int kc = 0; kc < 4; ++kc) {
      const h16* arow = &As[wave * 16 + row16][kc * 32 + 16 * hlf];
      v8h lo = *(const v8h*)arow;
      v8h hi = *(const v8h*)(arow + 8);
      afrag[kc] = __builtin_shufflevector(lo, hi, 0, 1, 2, 3, 4, 5, 6, 7,
                                          8, 9, 10, 11, 12, 13, 14, 15);
    }
    // kc-outer / nt-inner: consecutive WMMAs use different accumulators
#pragma unroll
    for (int kc = 0; kc < 4; ++kc) {
#pragma unroll
      for (int nt = 0; nt < 8; ++nt) {
        const h16* bp = WsF + (nt * 4 + kc) * WS_FRAG_STRIDE + myslot;
        v8h blo = *(const v8h*)bp;
        v8h bhi = *(const v8h*)(bp + 8);
        v16h bfrag = __builtin_shufflevector(blo, bhi, 0, 1, 2, 3, 4, 5, 6, 7,
                                             8, 9, 10, 11, 12, 13, 14, 15);
        acc[nt] = __builtin_amdgcn_wmma_f32_16x16x32_f16(
            false, afrag[kc], false, bfrag, (short)0, acc[nt], false, false);
      }
    }
    __syncthreads();
  }

#pragma unroll
  for (int nt = 0; nt < 8; ++nt) {
    int col = nt * 16 + row16;
#pragma unroll
    for (int r = 0; r < 8; ++r) {
      int grow = blockRow + wave * 16 + r + 8 * hlf;
      if (grow < M) C[(size_t)grow * ldc + col] = acc[nt][r];
    }
  }
}

// -------------------------------- host orchestration -----------------------

static inline int cdiv(long long a, long long b) { return (int)((a + b - 1) / b); }

extern "C" void kernel_launch(void* const* d_in, const int* in_sizes, int n_in,
                              void* d_out, int out_size, void* d_ws, size_t ws_size,
                              hipStream_t stream) {
  (void)in_sizes; (void)n_in; (void)out_size; (void)ws_size;
  auto F = [&](int i) { return (const float*)d_in[i]; };
  auto I = [&](int i) { return (const int*)d_in[i]; };

  // input leaf indices (pytree order of setup_inputs dict)
  // 0:x 1:pos_enc 2:node_id 3:z_initial
  // 4..9: z_params (g1,be1,zW,zb,g2,be2)
  // 10..17: xemb (W1,b1,g1,be1,W2,b2,g2,be2)
  // 18: conv1_eps 19:conv1_lin.W 20:conv1_lin.b
  // 21..28: conv1_nn
  // 29: convs_eps[2]  30..33: convs_lin (W0,b0,W1,b1)
  // 34..41: convs_nn[0]  42..49: convs_nn[1]
  // 50..55: head (W1,b1,gh,bh,W2,b2)
  // 56: edge_index[2,E] 57: batch 58: pos_index 59: pos_batch
  const int* src = I(56);
  const int* dst = I(56) + EE;

  // ---- workspace carve ----
  char* base = (char*)d_ws;
  size_t off = 0;
  auto carve = [&](size_t bytes) { void* p = base + off; off += (bytes + 255) & ~(size_t)255; return p; };
  float* edgeF   = (float*)carve((size_t)EE * HH * 4);   // z / edge GEMM outputs
  h16*   zemb16  = (h16*)  carve((size_t)EE * HH * 2);   // also holds bn(z) pre-GEMM
  h16*   nid16   = (h16*)  carve((size_t)EE * HH * 2);
  float* cat     = (float*)carve((size_t)NN * 512 * 4);  // 4 concat slices
  float* hmid    = (float*)carve((size_t)NN * HH * 4);
  float* aggrA   = (float*)carve((size_t)NN * HH * 4);
  float* aggrB   = (float*)carve((size_t)NN * HH * 4);
  float* nodeY   = (float*)carve((size_t)NN * HH * 4);
  h16*   u16     = (h16*)  carve((size_t)NN * HH * 2);
  h16*   h16buf  = (h16*)  carve((size_t)NN * HH * 2);
  float* aggrS   = (float*)carve((size_t)NN * 4);
  float* uS      = (float*)carve((size_t)NN * 4);
  h16*   wswz    = (h16*)  carve((size_t)4 * WS_HALVES * 2);  // up to K=512 swizzled
  float* stats   = (float*)carve(256 * 4);
  float* ss      = (float*)carve(256 * 4);
  float* csum    = (float*)carve(HH * 4);
  float* pooled  = (float*)carve((size_t)GG * 512 * 4);
  h16*   pooled16= (h16*)  carve((size_t)GG * 512 * 2);
  float* cnt     = (float*)carve(GG * 4);

  auto zero = [&](float* p, long long n) {
    k_zero_f32<<<cdiv(n, 256 * 8), 256, 0, stream>>>(p, n);
  };
  auto cvt16 = [&](const float* s, h16* d, long long n) {
    k_f32_to_f16<<<cdiv(n, 256 * 8), 256, 0, stream>>>(s, d, n);
  };
  auto bnorm = [&](const float* y, int rows, const float* g, const float* be) {
    zero(stats, 256);
    k_bn_stats<<<512, 128, 0, stream>>>(y, rows, stats);
    k_bn_finalize<<<1, 128, 0, stream>>>(stats, g, be, 1.f / (float)rows, ss);
  };
  auto apply = [&](const float* y, float* o32, int ldo, int acc, h16* o16, int rows) {
    k_bn_relu_apply<<<cdiv((long long)rows * HH, 256 * 4), 256, 0, stream>>>(
        y, ss, o32, ldo, acc, o16, rows);
  };
  auto gemm = [&](const h16* Ain, const float* Wf32, int K, const float* bias,
                  float* Cout, int M, int ldc) {
    k_w_swizzle<<<cdiv((long long)K * HH, 256), 256, 0, stream>>>(Wf32, wswz, K);
    k_gemm_wmma<<<cdiv(M, 64), 128, 0, stream>>>(Ain, wswz, bias, Cout, M, K, ldc);
  };
  // mlp2: two [M,128]x[128,128] WMMA GEMMs with fused BN+ReLU
  auto mlp2 = [&](const h16* uin, int M, int pbase, float* out32, int ldo, int acc) {
    gemm(uin, F(pbase + 0), HH, F(pbase + 1), nodeY, M, HH);
    bnorm(nodeY, M, F(pbase + 2), F(pbase + 3));
    apply(nodeY, nullptr, 0, 0, h16buf, M);
    gemm(h16buf, F(pbase + 4), HH, F(pbase + 5), nodeY, M, HH);
    bnorm(nodeY, M, F(pbase + 6), F(pbase + 7));
    apply(nodeY, out32, ldo, acc, nullptr, M);
  };
  // din==1 mlp2 (first layer is rank-1)
  auto mlp2_rank1 = [&](const float* uscalar, int M, int pbase, float* out32, int ldo) {
    k_rank1<<<cdiv((long long)M * HH, 256 * 4), 256, 0, stream>>>(
        uscalar, F(pbase + 0), F(pbase + 1), nodeY, M);
    bnorm(nodeY, M, F(pbase + 2), F(pbase + 3));
    apply(nodeY, nullptr, 0, 0, h16buf, M);
    gemm(h16buf, F(pbase + 4), HH, F(pbase + 5), nodeY, M, HH);
    bnorm(nodeY, M, F(pbase + 6), F(pbase + 7));
    apply(nodeY, out32, ldo, 0, nullptr, M);
  };

  // ================= phase Z : structural edge embedding =================
  zero(edgeF, (long long)EE * HH);
  k_scatter_pos<<<PP, 128, 0, stream>>>(F(3), F(1), I(58), I(59), edgeF, PP);
  bnorm(edgeF, EE, F(4), F(5));
  apply(edgeF, nullptr, 0, 0, zemb16, EE);                    // bn(z) -> f16
  gemm(zemb16, F(6), HH, F(7), edgeF, EE, HH);                // h @ zW + zb
  bnorm(edgeF, EE, F(8), F(9));
  apply(edgeF, nullptr, 0, 0, zemb16, EE);                    // z_emb (f16, persists)

  cvt16(F(2), nid16, (long long)EE * HH);                     // node_id -> f16

  // ================= xemb -> cat slice 0 =================
  mlp2_rank1(F(0), NN, 10, cat + 0, 512);

  // ================= conv1 -> cat slice 1 =================
  zero(aggrS, NN);
  k_conv1_edge<<<cdiv(EE, 8), 256, 0, stream>>>(zemb16, F(19), F(20), F(0), src, dst, aggrS, EE);
  k_combine_scalar<<<cdiv(NN, 256), 256, 0, stream>>>(F(0), aggrS, F(18), uS, NN);
  mlp2_rank1(uS, NN, 21, cat + HH, 512);

  // ================= conv layers =================
  for (int i = 0; i < 2; ++i) {
    const float* Wl = F(30 + 2 * i);
    const float* bl = F(31 + 2 * i);
    const float* epsp = F(29) + i;
    int nnb = 34 + 8 * i;
    const float* hprev = cat + (size_t)HH * (1 + i);          // ld 512
    float* outsl = cat + (size_t)HH * (2 + i);                // ld 512

    // --- GINE #1: z_emb edges ---
    gemm(zemb16, Wl, HH, bl, edgeF, EE, HH);
    zero(aggrA, (long long)NN * HH);
    k_edge_msg<<<cdiv((long long)EE * 32, 256), 256, 0, stream>>>(
        hprev, 512, edgeF, src, dst, aggrA, EE);
    k_combine_u<<<cdiv((long long)NN * HH, 256 * 4), 256, 0, stream>>>(
        hprev, 512, aggrA, epsp, u16, NN);
    mlp2(u16, NN, nnb, hmid, HH, 0);

    // --- GINE #2 + #3: node_id and node_id+1 (share GEMM via colsum) ---
    gemm(nid16, Wl, HH, bl, edgeF, EE, HH);
    k_colsum<<<1, 128, 0, stream>>>(Wl, csum);
    zero(aggrA, (long long)NN * HH);
    zero(aggrB, (long long)NN * HH);
    k_edge_msg_dual<<<cdiv((long long)EE * 32, 256), 256, 0, stream>>>(
        hmid, HH, edgeF, csum, src, dst, aggrA, aggrB, EE);
    k_combine_u<<<cdiv((long long)NN * HH, 256 * 4), 256, 0, stream>>>(
        hmid, HH, aggrA, epsp, u16, NN);
    mlp2(u16, NN, nnb, outsl, 512, 0);                        // write slice
    k_combine_u<<<cdiv((long long)NN * HH, 256 * 4), 256, 0, stream>>>(
        hmid, HH, aggrB, epsp, u16, NN);
    mlp2(u16, NN, nnb, outsl, 512, 1);                        // accumulate slice
  }

  // ================= mean pool =================
  zero(pooled, (long long)GG * 512);
  zero(cnt, GG);
  k_pool_add<<<NN, 128, 0, stream>>>(cat, I(57), pooled, cnt, NN);
  k_pool_div<<<cdiv(GG * 512, 256), 256, 0, stream>>>(pooled, cnt, pooled16);

  // ================= head =================
  gemm(pooled16, F(50), 512, F(51), nodeY, GG, HH);           // K = 512
  bnorm(nodeY, GG, F(52), F(53));
  apply(nodeY, nullptr, 0, 0, h16buf, GG);
  gemm(h16buf, F(54), HH, F(55), (float*)d_out, GG, OUTD);    // final logits
}
